// DetectionHardMinedCELoss_38508676776135
// MI455X (gfx1250) — compile-verified, matrix operand order checked by reading, not run on hardware
//
#include <hip/hip_runtime.h>
#include <hip/hip_bf16.h>

// DetectionHardMinedCELoss for MI455X (gfx1250, wave32).
// Kernel 1: streaming log-softmax CE. Per wave: 32 positions x 81 classes
//   staged in LDS (stride 33 floats -> conflict-free), column max in registers,
//   exp in f32 VALU, packed to f16, class-sum via V_WMMA_F32_16X16X32_F16 with
//   an all-ones A matrix (D[m][n] = sum_k B[k][n]); f32 accumulation.
// Kernel 2: per-batch hard-negative mining: exact top-k sum via bitwise
//   k-th-largest selection on float bit patterns (valid: losses >= 0).

typedef __attribute__((ext_vector_type(16))) _Float16 v16h;
typedef __attribute__((ext_vector_type(8)))  float    v8f;

constexpr int kB = 64;
constexpr int kC = 81;
constexpr int kP = 8732;
constexpr int kRowStride = 33;           // 32 cols + 1 pad (floats)

__global__ __launch_bounds__(128) void ce_loss_kernel(
    const float* __restrict__ logits,    // [B, C, P]
    const int*   __restrict__ labels,    // [B, P]
    float*       __restrict__ loss_out)  // [B, P] (workspace)
{
    __shared__ float tile[4][kC * kRowStride];   // 42,768 B

    const int wave = threadIdx.x >> 5;
    const int lane = threadIdx.x & 31;
    const int b    = blockIdx.y;
    const int p0   = (blockIdx.x * 4 + wave) * 32;

    float* t = tile[wave];

    // ---- Stage logits tile (coalesced 128B rows) + per-column running max ----
    const int pcol = min(p0 + lane, kP - 1);     // clamp tail (results discarded)
    const float* src = logits + (size_t)b * kC * kP + pcol;

    float maxv = -3.4e38f;
    #pragma unroll 3
    for (int r = 0; r < kC; ++r) {
        float v = src[(size_t)r * kP];
        t[r * kRowStride + lane] = v;
        maxv = fmaxf(maxv, v);
    }
    __syncthreads();   // make the wave's LDS tile visible across its lanes

    // ---- Build B operands (32x16 f16, n = lane%16, K split by lane half) ----
    const int halfk = lane >> 4;        // 0: K 0..15 of the k-tile, 1: K 16..31
    const int nidx  = lane & 15;        // column within each 16-position n-tile
    const float m0 = __shfl(maxv, nidx, 32);        // max of column nidx
    const float m1 = __shfl(maxv, 16 + nidx, 32);   // max of column 16+nidx

    v16h ones;
    #pragma unroll
    for (int i = 0; i < 16; ++i) ones[i] = (_Float16)1.0f;

    v8f acc0 = {};   // positions p0 .. p0+15
    v8f acc1 = {};   // positions p0+16 .. p0+31

    #pragma unroll
    for (int j = 0; j < 3; ++j) {                // K tiles: classes 32j..32j+31
        const int cbase = j * 32 + halfk * 16;
        v16h b0, b1;
        #pragma unroll
        for (int i = 0; i < 16; ++i) {
            const int  c  = cbase + i;
            const bool ok = (c < kC);
            const int  cc = ok ? c : 0;          // clamped, branchless
            float e0 = __expf(t[cc * kRowStride + nidx]      - m0);
            float e1 = __expf(t[cc * kRowStride + 16 + nidx] - m1);
            b0[i] = (_Float16)(ok ? e0 : 0.0f);  // padded classes contribute 0
            b1[i] = (_Float16)(ok ? e1 : 0.0f);
        }
        // D = A(ones,16x32) x B(exp,32x16) + C  ->  per-n sum of 32 exps
        acc0 = __builtin_amdgcn_wmma_f32_16x16x32_f16(
                   false, ones, false, b0, (short)0, acc0, false, false);
        acc1 = __builtin_amdgcn_wmma_f32_16x16x32_f16(
                   false, ones, false, b1, (short)0, acc1, false, false);
    }

    // C/D layout: VGPR0 holds (M=0,N=lane) for lanes 0-15 and (M=8,N=lane-16)
    // for lanes 16-31; all rows identical (A all ones) -> per-lane select.
    const float sumexp = (lane < 16) ? acc0[0] : acc1[0];
    const float lse    = maxv + __logf(sumexp);

    const int p = p0 + lane;
    if (p < kP) {
        const int   tc = labels[(size_t)b * kP + p];     // 0..80
        const float xt = t[tc * kRowStride + lane];      // lane == its column
        loss_out[(size_t)b * kP + p] = lse - xt;
    }
}

__global__ __launch_bounds__(256) void hardmine_kernel(
    const float* __restrict__ loss,      // [B, P]
    const int*   __restrict__ labels,    // [B, P]
    float*       __restrict__ out)       // [B]
{
    __shared__ float s_con[kP];          // 34,928 B
    __shared__ float s_f[8];
    __shared__ int   s_i[8];

    const int b    = blockIdx.x;
    const int tid  = threadIdx.x;
    const int lane = tid & 31;
    const int wv   = tid >> 5;

    // ---- pos_sum / pos_cnt / con_neg ----
    float psum = 0.0f;
    int   pcnt = 0;
    for (int p = tid; p < kP; p += 256) {
        const float l   = loss[(size_t)b * kP + p];
        const bool  pos = labels[(size_t)b * kP + p] > 0;
        psum += pos ? l : 0.0f;
        pcnt += pos ? 1 : 0;
        s_con[p] = pos ? 0.0f : l;       // positives excluded from mining
    }
    #pragma unroll
    for (int o = 16; o > 0; o >>= 1) {
        psum += __shfl_down(psum, o, 32);
        pcnt += __shfl_down(pcnt, o, 32);
    }
    if (lane == 0) { s_f[wv] = psum; s_i[wv] = pcnt; }
    __syncthreads();

    float tot_psum = 0.0f;
    int   tot_pcnt = 0;
    #pragma unroll
    for (int w = 0; w < 8; ++w) { tot_psum += s_f[w]; tot_pcnt += s_i[w]; }
    const int k = min(3 * tot_pcnt, kP);
    __syncthreads();

    // ---- bitwise k-th-largest: thr = largest c with count(bits >= c) >= k ----
    // Valid since all losses are >= 0 (max >= x_t and log(sumexp) >= 0), and
    // IEEE bit patterns of non-negative floats are order-isomorphic to uints.
    unsigned thr = 0u;
    if (k > 0) {
        for (int bit = 30; bit >= 0; --bit) {
            const unsigned c = thr | (1u << bit);
            int cnt = 0;
            for (int p = tid; p < kP; p += 256)
                cnt += (__float_as_uint(s_con[p]) >= c) ? 1 : 0;
            #pragma unroll
            for (int o = 16; o > 0; o >>= 1) cnt += __shfl_down(cnt, o, 32);
            if (lane == 0) s_i[wv] = cnt;
            __syncthreads();
            int total = 0;
            #pragma unroll
            for (int w = 0; w < 8; ++w) total += s_i[w];
            if (total >= k) thr = c;     // uniform decision across block
            __syncthreads();
        }
    }

    // ---- exact top-k sum: strict-greater sum + tie fill at thr ----
    float ts  = 0.0f;
    int   cgt = 0;
    for (int p = tid; p < kP; p += 256) {
        const float v = s_con[p];
        if (__float_as_uint(v) > thr) { ts += v; ++cgt; }
    }
    #pragma unroll
    for (int o = 16; o > 0; o >>= 1) {
        ts  += __shfl_down(ts, o, 32);
        cgt += __shfl_down(cgt, o, 32);
    }
    if (lane == 0) { s_f[wv] = ts; s_i[wv] = cgt; }
    __syncthreads();

    if (tid == 0) {
        float tts  = 0.0f;
        int   tcgt = 0;
        #pragma unroll
        for (int w = 0; w < 8; ++w) { tts += s_f[w]; tcgt += s_i[w]; }
        float r = tot_psum;
        if (k > 0)
            r += tts + (float)(k - tcgt) * __uint_as_float(thr);
        out[b] = r;
    }
}

extern "C" void kernel_launch(void* const* d_in, const int* in_sizes, int n_in,
                              void* d_out, int out_size, void* d_ws, size_t ws_size,
                              hipStream_t stream) {
    // setup_inputs order: pred_loc[0], pred_bclass[1], true_loc_vec[2], true_bclass[3]
    const float* pred_bclass = (const float*)d_in[1];
    const int*   true_bclass = (const int*)d_in[3];
    float*       out         = (float*)d_out;
    float*       ws_loss     = (float*)d_ws;     // B*P floats = 2,235,392 B

    (void)in_sizes; (void)n_in; (void)out_size; (void)ws_size;

    dim3 grid1((kP + 127) / 128, kB);            // 69 x 64 blocks, 4 waves each
    ce_loss_kernel<<<grid1, 128, 0, stream>>>(pred_bclass, true_bclass, ws_loss);
    hardmine_kernel<<<dim3(kB), 256, 0, stream>>>(ws_loss, true_bclass, out);
}
